// CrossAttentionValueFuser_11287174054082
// MI455X (gfx1250) — matrix operand order repaired and checked by reading, not compiled, not run on hardware
//
#include <hip/hip_runtime.h>

// ---------------------------------------------------------------------------
// CrossAttentionValueFuser for MI455X (gfx1250), wave32 + WMMA bf16 + TDM.
// ~300 GFLOP vs ~150MB HBM traffic -> matrix-core bound. All GEMMs run on
// v_wmma_f32_16x16x32_bf16; attention is flash-style (online softmax) with
// Tensor-Data-Mover double buffering (tensor_load_to_lds + s_wait_tensorcnt).
// ---------------------------------------------------------------------------

constexpr int BN   = 16;     // B*N
constexpr int NREP = 4;      // N (flow broadcast factor)
constexpr int PP   = 2304;   // H*W
constexpr int CXc  = 512;
constexpr int CFc  = 256;
constexpr int HIDc = 256;
constexpr int OUTc = 512;
constexpr int KTc  = 1536;   // 2*(CX+CF)

typedef __attribute__((ext_vector_type(16))) __bf16 v16bf;
typedef __attribute__((ext_vector_type(8)))  float  v8f;
typedef __attribute__((ext_vector_type(4)))  unsigned int u32x4;
typedef __attribute__((ext_vector_type(8)))  int i32x8;
typedef __attribute__((ext_vector_type(4)))  int i32x4;

__device__ __forceinline__ v8f vzero() {
    v8f z;
#pragma unroll
    for (int i = 0; i < 8; ++i) z[i] = 0.0f;
    return z;
}

__device__ __forceinline__ v8f wmma_bf16(v16bf a, v16bf b, v8f c) {
    return __builtin_amdgcn_wmma_f32_16x16x32_bf16(
        false, a, false, b, (short)0, c, false, false);
}

// A-fragment (16x32 bf16) from LDS row-major [m][k], stride in elems.
__device__ __forceinline__ v16bf frag_a(const __bf16* base, int stride, int lane) {
    int row = lane & 15;
    int c0  = (lane & 16) ? 8 : 0;
    union { v16bf v; uint4 q[2]; } u;
    u.q[0] = *(const uint4*)(base + row * stride + c0);
    u.q[1] = *(const uint4*)(base + row * stride + c0 + 16);
    return u.v;
}

// B-fragment (32x16 bf16) from LDS [n][k] (k contiguous), stride in elems.
__device__ __forceinline__ v16bf frag_b(const __bf16* base, int stride, int lane) {
    int n  = lane & 15;
    int c0 = (lane & 16) ? 8 : 0;
    union { v16bf v; uint4 q[2]; } u;
    u.q[0] = *(const uint4*)(base + n * stride + c0);
    u.q[1] = *(const uint4*)(base + n * stride + c0 + 16);
    return u.v;
}

// ---------------------------------------------------------------------------
// TDM tile load: issue a Tensor-Data-Mover DMA (global -> LDS) for a tile of
// `rows` rows x `td0` bf16 elements, global row stride `stride` elems.
// D# bitfields per CDNA5 ISA 8.3/8.4. rows==1 && stride==td0 -> plain 1D copy.
// Caller: one wave only; completion via s_wait_tensorcnt.
// ---------------------------------------------------------------------------
__device__ __forceinline__ void tdm_load_tile(__bf16* lds, const __bf16* g,
                                              unsigned td0, unsigned rows,
                                              unsigned long long stride, int lane) {
#if __has_builtin(__builtin_amdgcn_tensor_load_to_lds)
    unsigned long long ga = (unsigned long long)g;
    unsigned lds_off = (unsigned)(unsigned long long)lds;   // LDS offset = low 32 bits of the address
    u32x4 g0;
    g0[0] = 1u;                                   // count=1, user descriptor
    g0[1] = lds_off;                              // lds_addr (bytes)
    g0[2] = (unsigned)(ga & 0xffffffffu);         // global_addr[31:0]
    g0[3] = (unsigned)((ga >> 32) & 0x01ffffffu) | (2u << 30);  // [56:32] | type=2
    unsigned long long s0 = stride;               // tensor_dim0_stride (elems)
    i32x8 g1;
    g1[0] = (int)(1u << 16);                      // wg_mask=0, data_size=1 (2B)
    g1[1] = (int)((td0 & 0xffffu) << 16);         // abar=0 | tensor_dim0[15:0]
    g1[2] = (int)((td0 >> 16) | ((rows & 0xffffu) << 16));      // td0 hi | td1 lo
    g1[3] = (int)((rows >> 16) | ((td0 & 0xffffu) << 16));      // td1 hi | tile_dim0
    g1[4] = (int)(rows & 0xffffu);                // tile_dim1 | tile_dim2=0
    g1[5] = (int)(s0 & 0xffffffffu);              // stride0[31:0]
    g1[6] = (int)((s0 >> 32) & 0xffffu);          // stride0[47:32] | stride1 lo=0
    g1[7] = 0;
    i32x4 z4;
#pragma unroll
    for (int i = 0; i < 4; ++i) z4[i] = 0;
#if __clang_major__ >= 23
    i32x8 z8;
#pragma unroll
    for (int i = 0; i < 8; ++i) z8[i] = 0;
    __builtin_amdgcn_tensor_load_to_lds(g0, g1, z4, z4, z8, 0);
#else
    __builtin_amdgcn_tensor_load_to_lds(g0, g1, z4, z4, 0);
#endif
#else
    // Fallback: cooperative copy by the issuing wave (32 lanes).
    for (unsigned r = 0; r < rows; ++r)
        for (unsigned i = (unsigned)lane * 8; i < td0; i += 256)
            *(uint4*)(lds + (size_t)r * td0 + i) =
                *(const uint4*)(g + (size_t)r * stride + i);
#endif
}

template <int N>
__device__ __forceinline__ void tdm_wait() {
#if __has_builtin(__builtin_amdgcn_s_wait_tensorcnt)
    __builtin_amdgcn_s_wait_tensorcnt(N);   // constant-folded immediate
#endif
}

// ---------------------------------------------------------------------------
// Packing kernels: fp32 -> bf16 conversions / transposes
// ---------------------------------------------------------------------------
__global__ void cvt_f32_bf16(const float* __restrict__ src,
                             __bf16* __restrict__ dst, int n) {
    int i = blockIdx.x * blockDim.x + threadIdx.x;
    if (i < n) dst[i] = (__bf16)src[i];
}

__global__ void pack_mv(const float* __restrict__ mv,
                        __bf16* __restrict__ mv_h, __bf16* __restrict__ mvT) {
    size_t idx = (size_t)blockIdx.x * blockDim.x + threadIdx.x;
    size_t total = (size_t)BN * CXc * PP;
    if (idx >= total) return;
    int p  = (int)(idx % PP);
    int c  = (int)((idx / PP) % CXc);
    int bn = (int)(idx / ((size_t)PP * CXc));
    __bf16 h = (__bf16)mv[idx];
    mv_h[idx] = h;
    mvT[((size_t)bn * PP + p) * CXc + c] = h;
}

__global__ void pack_ff(const float* __restrict__ ff,
                        __bf16* __restrict__ ff_h, __bf16* __restrict__ ffT) {
    size_t idx = (size_t)blockIdx.x * blockDim.x + threadIdx.x;
    size_t total = (size_t)BN * CFc * PP;
    if (idx >= total) return;
    int p  = (int)(idx % PP);
    int c  = (int)((idx / PP) % CFc);
    int bn = (int)(idx / ((size_t)PP * CFc));
    int b  = bn / NREP;
    __bf16 h = (__bf16)ff[((size_t)b * CFc + c) * PP + p];
    ff_h[idx] = h;
    ffT[((size_t)bn * PP + p) * CFc + c] = h;
}

// ---------------------------------------------------------------------------
// Projection GEMM: Out[bn][PP][HID] = XT[bn][PP][Kd] * W[HID][Kd]^T + bias[h]
// ---------------------------------------------------------------------------
__global__ __launch_bounds__(128) void gemm_proj(
    const __bf16* __restrict__ XT, const __bf16* __restrict__ W,
    const float* __restrict__ bias, __bf16* __restrict__ Out, int Kd) {
    __shared__ __bf16 lA[64 * 32];
    __shared__ __bf16 lB[64 * 32];
    int bn = blockIdx.z;
    int p0 = blockIdx.x * 64;
    int h0 = blockIdx.y * 64;
    int t = threadIdx.x, wave = t >> 5, lane = t & 31;
    int lane16 = lane & 15, rbase = (lane & 16) ? 8 : 0;
    const __bf16* Xb = XT + (size_t)bn * PP * Kd;

    v8f acc[4];
#pragma unroll
    for (int i = 0; i < 4; ++i) acc[i] = vzero();

    int rowA = t >> 1, halfA = t & 1;
    for (int k0 = 0; k0 < Kd; k0 += 32) {
        __syncthreads();
        {
            const __bf16* sa = Xb + (size_t)(p0 + rowA) * Kd + k0 + halfA * 16;
            uint4* da = (uint4*)(lA + rowA * 32 + halfA * 16);
            da[0] = ((const uint4*)sa)[0];
            da[1] = ((const uint4*)sa)[1];
            if (k0 + 32 < Kd) __builtin_prefetch(sa + 32, 0, 3);
            const __bf16* sb = W + (size_t)(h0 + rowA) * Kd + k0 + halfA * 16;
            uint4* db = (uint4*)(lB + rowA * 32 + halfA * 16);
            db[0] = ((const uint4*)sb)[0];
            db[1] = ((const uint4*)sb)[1];
            if (k0 + 32 < Kd) __builtin_prefetch(sb + 32, 0, 3);
        }
        __syncthreads();
        v16bf af = frag_a(lA + wave * 16 * 32, 32, lane);
#pragma unroll
        for (int ns = 0; ns < 4; ++ns) {
            v16bf bf = frag_b(lB + ns * 16 * 32, 32, lane);
            acc[ns] = wmma_bf16(af, bf, acc[ns]);
        }
    }

    __bf16* Ob = Out + (size_t)bn * PP * HIDc;
#pragma unroll
    for (int ns = 0; ns < 4; ++ns) {
        int h = h0 + ns * 16 + lane16;
        float bv = bias[h];
#pragma unroll
        for (int r = 0; r < 8; ++r) {
            int p = p0 + wave * 16 + r + rbase;
            Ob[(size_t)p * HIDc + h] = (__bf16)(acc[ns][r] + bv);
        }
    }
}

// ---------------------------------------------------------------------------
// Flash cross-attention with TDM double buffering.
// Qt,Kt: [bn][PP][HID] bf16.  V: [bn][Cv][PP] bf16.  Ow: [bn][Cv][PP] bf16.
// grid = (PP/64, Cv/256, bn); 128 threads / 4 waves; key chunks of 64.
// LDS: lQ 32K | lK x2 64K | lV x2 64K | lP 8K = 168KB dynamic.
// ---------------------------------------------------------------------------
__global__ __launch_bounds__(128) void attn_kernel(
    const __bf16* __restrict__ Qt, const __bf16* __restrict__ Kt,
    const __bf16* __restrict__ V, __bf16* __restrict__ Ow, int Cv) {
    extern __shared__ __bf16 smem[];
    __bf16* lQ  = smem;                  // 64 x 256
    __bf16* lK0 = lQ + 64 * 256;         // 64 x 256 (buf 0)   [q][h]
    __bf16* lK1 = lK0 + 64 * 256;        // 64 x 256 (buf 1)
    __bf16* lV0 = lK1 + 64 * 256;        // 256 x 64 (buf 0)   [c][q]
    __bf16* lV1 = lV0 + 256 * 64;        // 256 x 64 (buf 1)
    __bf16* lP  = lV1 + 256 * 64;        // 4 waves x 16 x 64  (P-bar)

    int bn = blockIdx.z;
    int m0 = blockIdx.x * 64;
    int c0 = blockIdx.y * 256;
    int t = threadIdx.x, wave = t >> 5, lane = t & 31;
    int lane16 = lane & 15, rbase = (lane & 16) ? 8 : 0;

    const __bf16* Qb = Qt + (size_t)bn * PP * HIDc;
    const __bf16* Kb = Kt + (size_t)bn * PP * HIDc;
    const __bf16* Vb = V + (size_t)bn * Cv * PP + (size_t)c0 * PP;

    constexpr int NCH = PP / 64;   // 36 key chunks

    // Prologue: DMA the query tile and first K/V chunk.
    if (wave == 0) {
        tdm_load_tile(lQ, Qb + (size_t)m0 * HIDc, 64 * 256, 1, 64 * 256, lane);
        tdm_load_tile(lK0, Kb, 64 * 256, 1, 64 * 256, lane);
        tdm_load_tile(lV0, Vb, 64, 256, PP, lane);
    }

    v8f Oacc[16];
#pragma unroll
    for (int i = 0; i < 16; ++i) Oacc[i] = vzero();
    float m_s[8], l_s[8];
#pragma unroll
    for (int r = 0; r < 8; ++r) { m_s[r] = -1.0e30f; l_s[r] = 0.0f; }

    for (int ic = 0; ic < NCH; ++ic) {
        const __bf16* cK = (ic & 1) ? lK1 : lK0;
        const __bf16* cV = (ic & 1) ? lV1 : lV0;
        __bf16* nK = (ic & 1) ? lK0 : lK1;
        __bf16* nV = (ic & 1) ? lV0 : lV1;

        __syncthreads();   // next-buffers free for DMA overwrite
        if (wave == 0) {
            if (ic + 1 < NCH) {
                int qn = (ic + 1) * 64;
                tdm_load_tile(nK, Kb + (size_t)qn * HIDc, 64 * 256, 1, 64 * 256, lane);
                tdm_load_tile(nV, Vb + qn, 64, 256, PP, lane);
                tdm_wait<2>();   // current chunk (and Q) landed; next in flight
            } else {
                tdm_wait<0>();
            }
        }
        __syncthreads();   // current buffers ready for all waves

        // S(16x64) = Q(16x256) * K_chunk^T : 32 WMMAs per wave
        v8f S[4];
#pragma unroll
        for (int qs = 0; qs < 4; ++qs) S[qs] = vzero();
#pragma unroll
        for (int kk = 0; kk < 8; ++kk) {
            v16bf af = frag_a(lQ + wave * 16 * 256 + kk * 32, 256, lane);
#pragma unroll
            for (int qs = 0; qs < 4; ++qs) {
                v16bf bf = frag_b(cK + qs * 16 * 256 + kk * 32, 256, lane);
                S[qs] = wmma_bf16(af, bf, S[qs]);
            }
        }

        // online softmax (rows on (vgpr r, lane-half), cols on 16 lanes)
        float cm[8], sc[8], rs[8], pb[4][8];
#pragma unroll
        for (int r = 0; r < 8; ++r)
            cm[r] = fmaxf(fmaxf(S[0][r], S[1][r]), fmaxf(S[2][r], S[3][r]));
#pragma unroll
        for (int m = 1; m < 16; m <<= 1)
#pragma unroll
            for (int r = 0; r < 8; ++r)
                cm[r] = fmaxf(cm[r], __shfl_xor(cm[r], m, 32));
#pragma unroll
        for (int r = 0; r < 8; ++r) {
            float mn = fmaxf(m_s[r], cm[r]);
            sc[r] = __expf(m_s[r] - mn);
            m_s[r] = mn;
            rs[r] = 0.0f;
#pragma unroll
            for (int qs = 0; qs < 4; ++qs) {
                pb[qs][r] = __expf(S[qs][r] - mn);
                rs[r] += pb[qs][r];
            }
        }
#pragma unroll
        for (int m = 1; m < 16; m <<= 1)
#pragma unroll
            for (int r = 0; r < 8; ++r) rs[r] += __shfl_xor(rs[r], m, 32);
#pragma unroll
        for (int r = 0; r < 8; ++r) l_s[r] = l_s[r] * sc[r] + rs[r];
#pragma unroll
        for (int cs = 0; cs < 16; ++cs)
#pragma unroll
            for (int r = 0; r < 8; ++r) Oacc[cs][r] *= sc[r];

        // D-layout -> A-layout via per-wave LDS round trip
        __bf16* pw = lP + wave * 16 * 64;
#pragma unroll
        for (int r = 0; r < 8; ++r)
#pragma unroll
            for (int qs = 0; qs < 4; ++qs)
                pw[(r + rbase) * 64 + qs * 16 + lane16] = (__bf16)pb[qs][r];
        __syncthreads();

        // O(16x256) += Pbar(16x64) * V_chunk^T : 32 WMMAs per wave
        v16bf pa0 = frag_a(pw, 64, lane);
        v16bf pa1 = frag_a(pw + 32, 64, lane);
#pragma unroll
        for (int cs = 0; cs < 16; ++cs) {
            Oacc[cs] = wmma_bf16(pa0, frag_b(cV + cs * 16 * 64, 64, lane), Oacc[cs]);
            Oacc[cs] = wmma_bf16(pa1, frag_b(cV + cs * 16 * 64 + 32, 64, lane), Oacc[cs]);
        }
    }

    __bf16* Ob = Ow + (size_t)bn * Cv * PP;
#pragma unroll
    for (int r = 0; r < 8; ++r) {
        float inv = 1.0f / l_s[r];
        int p = m0 + wave * 16 + r + rbase;
#pragma unroll
        for (int cs = 0; cs < 16; ++cs) {
            int c = c0 + cs * 16 + lane16;
            Ob[(size_t)c * PP + p] = (__bf16)(Oacc[cs][r] * inv);
        }
    }
}

// ---------------------------------------------------------------------------
// Fusion GEMM: out[bn][o][p] = sum_k wdr[o][k] * feat[k][p] + bdr[o]
// feat rows: [0,512)=mv, [512,1024)=wl, [1024,1280)=ff, [1280,1536)=wr
// ---------------------------------------------------------------------------
__global__ __launch_bounds__(128) void final_gemm(
    const __bf16* __restrict__ mv_h, const __bf16* __restrict__ wl_h,
    const __bf16* __restrict__ ff_h, const __bf16* __restrict__ wr_h,
    const __bf16* __restrict__ wdr_h, const float* __restrict__ bdr,
    float* __restrict__ out) {
    __shared__ __bf16 lA[64 * 32];
    __shared__ __bf16 lB[64 * 32];
    int bn = blockIdx.z;
    int p0 = blockIdx.x * 64;
    int o0 = blockIdx.y * 64;
    int t = threadIdx.x, wave = t >> 5, lane = t & 31;
    int lane16 = lane & 15, rbase = (lane & 16) ? 8 : 0;
    size_t bnCXP = (size_t)bn * CXc * PP;
    size_t bnCFP = (size_t)bn * CFc * PP;

    v8f acc[4];
#pragma unroll
    for (int i = 0; i < 4; ++i) acc[i] = vzero();

    int rowA = t >> 1, halfA = t & 1;
    for (int k0 = 0; k0 < KTc; k0 += 32) {
        __syncthreads();
        {
            const __bf16* sa = wdr_h + (size_t)(o0 + rowA) * KTc + k0 + halfA * 16;
            uint4* da = (uint4*)(lA + rowA * 32 + halfA * 16);
            da[0] = ((const uint4*)sa)[0];
            da[1] = ((const uint4*)sa)[1];
            if (k0 + 32 < KTc) __builtin_prefetch(sa + 32, 0, 3);
        }
        {
            int kg = k0 + (t >> 2);
            const __bf16* rowp;
            if (kg < 512)       rowp = mv_h + bnCXP + (size_t)kg * PP;
            else if (kg < 1024) rowp = wl_h + bnCXP + (size_t)(kg - 512) * PP;
            else if (kg < 1280) rowp = ff_h + bnCFP + (size_t)(kg - 1024) * PP;
            else                rowp = wr_h + bnCFP + (size_t)(kg - 1280) * PP;
            union { uint4 q[2]; __bf16 e[16]; } u;
            u.q[0] = *(const uint4*)(rowp + p0 + (t & 3) * 16);
            u.q[1] = *(const uint4*)(rowp + p0 + (t & 3) * 16 + 8);
            int kl = t >> 2, pl = (t & 3) * 16;
#pragma unroll
            for (int i = 0; i < 16; ++i) lB[(pl + i) * 32 + kl] = u.e[i];
        }
        __syncthreads();
        v16bf af = frag_a(lA + wave * 16 * 32, 32, lane);
#pragma unroll
        for (int ns = 0; ns < 4; ++ns) {
            v16bf bf = frag_b(lB + ns * 16 * 32, 32, lane);
            acc[ns] = wmma_bf16(af, bf, acc[ns]);
        }
    }

    float* Ob = out + (size_t)bn * OUTc * PP;
#pragma unroll
    for (int r = 0; r < 8; ++r) {
        int o = o0 + wave * 16 + r + rbase;
        float bv = bdr[o];
#pragma unroll
        for (int ns = 0; ns < 4; ++ns) {
            int p = p0 + ns * 16 + lane16;
            Ob[(size_t)o * PP + p] = acc[ns][r] + bv;
        }
    }
}

// ---------------------------------------------------------------------------
extern "C" void kernel_launch(void* const* d_in, const int* in_sizes, int n_in,
                              void* d_out, int out_size, void* d_ws, size_t ws_size,
                              hipStream_t stream) {
    const float* mv  = (const float*)d_in[0];
    const float* ff  = (const float*)d_in[1];
    const float* wq1 = (const float*)d_in[2];
    const float* bq1 = (const float*)d_in[3];
    const float* wk1 = (const float*)d_in[4];
    const float* bk1 = (const float*)d_in[5];
    const float* wq2 = (const float*)d_in[6];
    const float* bq2 = (const float*)d_in[7];
    const float* wk2 = (const float*)d_in[8];
    const float* bk2 = (const float*)d_in[9];
    const float* wdr = (const float*)d_in[10];
    const float* bdr = (const float*)d_in[11];
    float* out = (float*)d_out;

    __bf16* base = (__bf16*)d_ws;
    size_t off = 0;
    auto alloc = [&](size_t n) {
        __bf16* p = base + off;
        off += (n + 127) & ~(size_t)127;
        return p;
    };
    __bf16* mv_h = alloc((size_t)BN * CXc * PP);
    __bf16* mvT  = alloc((size_t)BN * PP * CXc);
    __bf16* ff_h = alloc((size_t)BN * CFc * PP);
    __bf16* ffT  = alloc((size_t)BN * PP * CFc);
    __bf16* q1t  = alloc((size_t)BN * PP * HIDc);
    __bf16* k1t  = alloc((size_t)BN * PP * HIDc);
    __bf16* q2t  = alloc((size_t)BN * PP * HIDc);
    __bf16* k2t  = alloc((size_t)BN * PP * HIDc);
    __bf16* wl_h = alloc((size_t)BN * CXc * PP);
    __bf16* wr_h = alloc((size_t)BN * CFc * PP);
    __bf16* wq1h = alloc((size_t)HIDc * CXc);
    __bf16* wk1h = alloc((size_t)HIDc * CFc);
    __bf16* wq2h = alloc((size_t)HIDc * CFc);
    __bf16* wk2h = alloc((size_t)HIDc * CXc);
    __bf16* wdrh = alloc((size_t)OUTc * KTc);

    auto cvt = [&](const float* s, __bf16* d, int n) {
        cvt_f32_bf16<<<(n + 255) / 256, 256, 0, stream>>>(s, d, n);
    };
    cvt(wq1, wq1h, HIDc * CXc);
    cvt(wk1, wk1h, HIDc * CFc);
    cvt(wq2, wq2h, HIDc * CFc);
    cvt(wk2, wk2h, HIDc * CXc);
    cvt(wdr, wdrh, OUTc * KTc);

    {
        size_t tot = (size_t)BN * CXc * PP;
        pack_mv<<<(unsigned)((tot + 255) / 256), 256, 0, stream>>>(mv, mv_h, mvT);
        size_t tof = (size_t)BN * CFc * PP;
        pack_ff<<<(unsigned)((tof + 255) / 256), 256, 0, stream>>>(ff, ff_h, ffT);
    }

    dim3 pg(PP / 64, HIDc / 64, BN);
    gemm_proj<<<pg, 128, 0, stream>>>(mvT, wq1h, bq1, q1t, CXc);  // Q1
    gemm_proj<<<pg, 128, 0, stream>>>(ffT, wk1h, bk1, k1t, CFc);  // K1
    gemm_proj<<<pg, 128, 0, stream>>>(ffT, wq2h, bq2, q2t, CFc);  // Q2
    gemm_proj<<<pg, 128, 0, stream>>>(mvT, wk2h, bk2, k2t, CXc);  // K2

    // attention: lQ + 2*lK + 2*lV + lP = 86016 bf16 elems = 168KB dynamic LDS
    size_t smem = (size_t)(64 * 256 * 3 + 256 * 64 * 2 + 4 * 16 * 64) * sizeof(__bf16);
    (void)hipFuncSetAttribute((const void*)attn_kernel,
                              hipFuncAttributeMaxDynamicSharedMemorySize,
                              (int)smem);
    dim3 a1(PP / 64, CFc / 256, BN);   // layer1: V = ff -> weighted_r
    attn_kernel<<<a1, 128, smem, stream>>>(q1t, k1t, ff_h, wr_h, CFc);
    dim3 a2(PP / 64, CXc / 256, BN);   // layer2: V = mv -> weighted_l
    attn_kernel<<<a2, 128, smem, stream>>>(q2t, k2t, mv_h, wl_h, CXc);

    dim3 fg(PP / 64, OUTc / 64, BN);
    final_gemm<<<fg, 128, 0, stream>>>(mv_h, wl_h, ff_h, wr_h, wdrh, bdr, out);
}
